// DeepHandwritingSynthesisExplicitModel_54855322305218
// MI455X (gfx1250) — matrix operand order repaired and checked by reading, not compiled, not run on hardware
//
#include <hip/hip_runtime.h>
#include <hip/hip_bf16.h>

typedef __attribute__((ext_vector_type(16))) _Float16 v16h;
typedef __attribute__((ext_vector_type(8)))  _Float16 v8h;
typedef __attribute__((ext_vector_type(8)))  float    v8f;

#define TSTEPS 600
#define ULEN   50
#define UNITS  400
#define NKMIX  10
#define NCHARS 73
#define NMIX   20

// canonical xin layout: [x(0..2) | h1(3..402) | w(403..475) | h2(476..875) | h3(876..1275)]
#define XW   1280
#define SH1  3
#define SW   403
#define SH2  476
#define SH3  876
#define K1   480
#define K2   896
#define K3   1280
#define KT1  15
#define KT2  28
#define KT3  40
#define KMDN 416
#define NMDN 128

__device__ __forceinline__ float sigf(float x) { return 1.f / (1.f + __expf(-x)); }

// WGP-scope prefetch (scope field 0 = WGP): a miss fills every cache level
// (CDNA5 ISA 10.5) — unlike __builtin_prefetch, which lowers to
// SE/SYS scope and never populates WGP$.
__device__ __forceinline__ void prefetch_wgp(const void* p) {
  asm volatile("global_prefetch_b8 %0, off" : : "v"(p));
}

// ---- WMMA fragment helpers (wave32, 16x16x32 f16 -> f32) -------------------
// A 16x32: lane m = lane&15; halves 0..15 -> K = k0 + (lane>>4)*16 .. +15.
__device__ __forceinline__ v16h load_a_frag(const _Float16* row, int kh) {
  v8h lo = *(const v8h*)(row + 2 * kh);
  v8h hi = *(const v8h*)(row + 2 * kh + 8);
  v16h a;
#pragma unroll
  for (int i = 0; i < 8; ++i) { a[i] = lo[i]; a[i + 8] = hi[i]; }
  return a;
}

#define WMMA(a, b, c) __builtin_amdgcn_wmma_f32_16x16x32_f16(false, (a), false, (b), \
                                                             (short)0, (c), false, false)

// Accumulate 4 gate tiles (i,f,g,o) of one unit-tile for one LSTM.
// Rolled k-loop (unroll 1): per iteration 4 WMMAs + 8 global_b128 + 2 ds_b128;
// WGP-scope prefetch one tile ahead turns next iteration's demand loads into
// WGP$ hits at zero VGPR cost. 25 resident waves hide the remaining latency.
__device__ __forceinline__ void lstm_mm(const _Float16* __restrict__ Wt, int kstride,
                                        int ktiles, const _Float16* __restrict__ xin,
                                        int m, int hi, int u, v8f acc[4]) {
  const _Float16* wr0 = Wt + (size_t)(0 * UNITS + u) * kstride + hi * 16;
  const _Float16* wr1 = Wt + (size_t)(1 * UNITS + u) * kstride + hi * 16;
  const _Float16* wr2 = Wt + (size_t)(2 * UNITS + u) * kstride + hi * 16;
  const _Float16* wr3 = Wt + (size_t)(3 * UNITS + u) * kstride + hi * 16;
#pragma unroll 1
  for (int kt = 0; kt < ktiles; ++kt) {
    const int k0 = kt * 32;
    v16h a = load_a_frag(xin + m * XW + k0, hi * 8);
    if (kt + 1 < ktiles) {
      const int kn = k0 + 32;
      prefetch_wgp(wr0 + kn);
      prefetch_wgp(wr1 + kn);
      prefetch_wgp(wr2 + kn);
      prefetch_wgp(wr3 + kn);
    }
    v16h b0 = *(const v16h*)(wr0 + k0);
    acc[0] = WMMA(a, b0, acc[0]);
    v16h b1 = *(const v16h*)(wr1 + k0);
    acc[1] = WMMA(a, b1, acc[1]);
    v16h b2 = *(const v16h*)(wr2 + k0);
    acc[2] = WMMA(a, b2, acc[2]);
    v16h b3 = *(const v16h*)(wr3 + k0);
    acc[3] = WMMA(a, b3, acc[3]);
  }
}

__device__ __forceinline__ void lstm_pointwise(v8f acc[4], v8f& c,
                                               float bi, float bf, float bg, float bo,
                                               float pi, float pf, float po, v8f& hout) {
#pragma unroll
  for (int r = 0; r < 8; ++r) {
    float cv = c[r];
    float ig = sigf(acc[0][r] + bi + pi * cv);
    float fg = sigf(acc[1][r] + bf + pf * cv);
    float gg = tanhf(acc[2][r] + bg);
    float cn = fg * cv + ig * gg;
    float og = sigf(acc[3][r] + bo + po * cn);
    c[r] = cn;
    hout[r] = og * tanhf(cn);
  }
}

__device__ __forceinline__ void store_h_lds(_Float16* xin, int base, int u, int hi, v8f h) {
#pragma unroll
  for (int r = 0; r < 8; ++r) {
    int m = r + hi * 8;
    xin[m * XW + base + u] = (_Float16)h[r];
  }
}

// ---- weight transpose/convert kernels (fp32 -> f16, canonical K layout) ----
__global__ void conv_w1(const float* __restrict__ W, _Float16* __restrict__ Wt) {
  int idx = blockIdx.x * blockDim.x + threadIdx.x;
  if (idx >= 1600 * K1) return;
  int n = idx / K1, kc = idx % K1;
  float v = 0.f;
  // orig lstm1 rows: [w(0..72), x(73..75), h1(76..475)] -> canonical [x,h1,w]
  if (kc < 403)      v = W[(kc + 73) * 1600 + n];   // x (73..75) + h1 (76..475)
  else if (kc < 476) v = W[(kc - 403) * 1600 + n];  // w (0..72)
  Wt[idx] = (_Float16)v;
}

__global__ void conv_w2(const float* __restrict__ W, _Float16* __restrict__ Wt) {
  int idx = blockIdx.x * blockDim.x + threadIdx.x;
  if (idx >= 1600 * K2) return;
  int n = idx / K2, k = idx % K2;
  Wt[idx] = (_Float16)((k < 876) ? W[k * 1600 + n] : 0.f);  // already canonical order
}

__global__ void conv_w3(const float* __restrict__ W, _Float16* __restrict__ Wt) {
  int idx = blockIdx.x * blockDim.x + threadIdx.x;
  if (idx >= 1600 * K3) return;
  int n = idx / K3, kc = idx % K3;
  float v = 0.f;
  // orig lstm3 rows: [x(0..2), h2(3..402), w(403..475), h3(476..875)]
  if (kc < 3)         v = W[kc * 1600 + n];            // x
  else if (kc < 403)  v = 0.f;                         // h1 region unused
  else if (kc < 476)  v = W[kc * 1600 + n];            // w (same slot)
  else if (kc < 876)  v = W[(kc - 473) * 1600 + n];    // h2
  else if (kc < 1276) v = W[(kc - 400) * 1600 + n];    // h3
  Wt[idx] = (_Float16)v;
}

__global__ void conv_wa(const float* __restrict__ W, _Float16* __restrict__ Wt) {
  int idx = blockIdx.x * blockDim.x + threadIdx.x;
  if (idx >= 32 * K1) return;
  int n = idx / K1, kc = idx % K1;
  float v = 0.f;
  // att_kernel orig rows: [w(0..72), x(73..75), h1(76..475)] -> canonical; pad N 30->32
  if (n < 30) {
    if (kc < 403)      v = W[(kc + 73) * 30 + n];
    else if (kc < 476) v = W[(kc - 403) * 30 + n];
  }
  Wt[idx] = (_Float16)v;
}

__global__ void conv_wm(const float* __restrict__ W, _Float16* __restrict__ Wt) {
  int idx = blockIdx.x * blockDim.x + threadIdx.x;
  if (idx >= NMDN * KMDN) return;
  int n = idx / KMDN, k = idx % KMDN;
  Wt[idx] = (_Float16)((n < 121 && k < 400) ? W[k * 121 + n] : 0.f);
}

// ---- persistent recurrent kernel: 16 blocks x 800 threads (25 waves) -------
__global__ __launch_bounds__(800) void handwriting_recurrent(
    const float* __restrict__ inputs, const int* __restrict__ char_seq,
    const int* __restrict__ char_len,
    const _Float16* __restrict__ W1t, const _Float16* __restrict__ W2t,
    const _Float16* __restrict__ W3t, const _Float16* __restrict__ Watt,
    const float* __restrict__ b1, const float* __restrict__ p1i,
    const float* __restrict__ p1f, const float* __restrict__ p1o,
    const float* __restrict__ b2, const float* __restrict__ p2i,
    const float* __restrict__ p2f, const float* __restrict__ p2o,
    const float* __restrict__ b3, const float* __restrict__ p3i,
    const float* __restrict__ p3f, const float* __restrict__ p3o,
    const float* __restrict__ attb, _Float16* __restrict__ hseq) {
  __shared__ __align__(16) _Float16 xin[16 * XW];   // 40 KB activation/state buffer
  __shared__ float wS[16 * 80];                     // soft window (padded 73->80)
  __shared__ float kappaS[16 * 16];                 // kappa (padded 10->16)
  __shared__ float attS[16 * 32];                   // exp(att logits) (padded 30->32)
  __shared__ int   lenS[16];
  __shared__ int   charS[16 * ULEN];

  const int tid  = threadIdx.x;
  const int lane = tid & 31;
  const int wid  = tid >> 5;            // 0..24 : one 16-unit tile per wave
  const int hi   = lane >> 4;           // half-wave index
  const int m    = lane & 15;           // A-frag row
  const int u    = wid * 16 + (lane & 15);  // unit column owned by this lane
  const int row0 = blockIdx.x * 16;

  // ---- init state ----
  for (int i = tid; i < 16 * XW; i += 800) xin[i] = (_Float16)0.f;
  if (tid < 16 * 16) kappaS[tid] = 0.f;
  if (tid < 16) lenS[tid] = char_len[row0 + tid];
  for (int i = tid; i < 16 * ULEN; i += 800)
    charS[i] = char_seq[(size_t)(row0 + i / ULEN) * ULEN + (i % ULEN)];

  // per-lane constant preloads (bias + peepholes for owned unit)
  float bi1 = b1[u], bf1 = b1[UNITS + u], bg1 = b1[2 * UNITS + u], bo1 = b1[3 * UNITS + u];
  float bi2 = b2[u], bf2 = b2[UNITS + u], bg2 = b2[2 * UNITS + u], bo2 = b2[3 * UNITS + u];
  float bi3 = b3[u], bf3 = b3[UNITS + u], bg3 = b3[2 * UNITS + u], bo3 = b3[3 * UNITS + u];
  float q1i = p1i[u], q1f = p1f[u], q1o = p1o[u];
  float q2i = p2i[u], q2f = p2f[u], q2o = p2o[u];
  float q3i = p3i[u], q3f = p3f[u], q3o = p3o[u];

  v8f c1 = {}, c2 = {}, c3 = {};   // cell states live in accumulator registers
  __syncthreads();

#pragma unroll 1
  for (int t = 0; t < TSTEPS; ++t) {
    // (A) load x_t into xin; clear window accumulator
    if (tid < 48) {
      int r = tid / 3, d = tid % 3;
      xin[r * XW + d] = (_Float16)inputs[((size_t)(row0 + r) * TSTEPS + t) * 3 + d];
    }
    for (int i = tid; i < 16 * 80; i += 800) wS[i] = 0.f;
    __syncthreads();

    // (B) LSTM1: reads [x, h1_old, w_old]
    {
      v8f acc[4] = {{}, {}, {}, {}};
      lstm_mm(W1t, K1, KT1, xin, m, hi, u, acc);
      v8f h;
      lstm_pointwise(acc, c1, bi1, bf1, bg1, bo1, q1i, q1f, q1o, h);
      __syncthreads();                 // everyone done reading h1_old
      store_h_lds(xin, SH1, u, hi, h);
    }
    __syncthreads();

    // (C) attention GEMM via WMMA on waves 0,1: [w_old,x,h1_new] @ attK (K=480, N=32)
    if (wid < 2) {
      const _Float16* wr = Watt + (size_t)(wid * 16 + (lane & 15)) * K1 + hi * 16;
      v8f za = {};
#pragma unroll 1
      for (int kt = 0; kt < KT1; ++kt) {
        v16h a = load_a_frag(xin + m * XW + kt * 32, hi * 8);
        v16h bb = *(const v16h*)(wr + kt * 32);
        za = WMMA(a, bb, za);
      }
      int col = wid * 16 + (lane & 15);
      float bv = (col < 30) ? attb[col] : 0.f;
#pragma unroll
      for (int r = 0; r < 8; ++r)
        attS[(r + hi * 8) * 32 + col] = __expf(za[r] + bv);
    }
    __syncthreads();
    if (tid < 160) {                  // kappa += dkappa
      int r = tid / NKMIX, j = tid % NKMIX;
      kappaS[r * 16 + j] += attS[r * 32 + 20 + j];
    }
    __syncthreads();
    {                                  // phi + one-hot window gather
      int r = tid / ULEN, uu = tid % ULEN;   // tid < 800 == 16*50 exactly
      if (uu < lenS[r]) {
        float phi = 0.f;
#pragma unroll
        for (int j = 0; j < NKMIX; ++j) {
          float d = kappaS[r * 16 + j] - (float)uu;
          phi += attS[r * 32 + j] * __expf(-attS[r * 32 + 10 + j] * d * d);
        }
        atomicAdd(&wS[r * 80 + charS[r * ULEN + uu]], phi);
      }
    }
    __syncthreads();
    for (int i = tid; i < 16 * NCHARS; i += 800) {  // new w -> xin
      int r = i / NCHARS, c = i % NCHARS;
      xin[r * XW + SW + c] = (_Float16)wS[r * 80 + c];
    }
    __syncthreads();

    // (D) LSTM2: reads [x, h1_new, w_new, h2_old]
    {
      v8f acc[4] = {{}, {}, {}, {}};
      lstm_mm(W2t, K2, KT2, xin, m, hi, u, acc);
      v8f h;
      lstm_pointwise(acc, c2, bi2, bf2, bg2, bo2, q2i, q2f, q2o, h);
      __syncthreads();
      store_h_lds(xin, SH2, u, hi, h);
    }
    __syncthreads();

    // (E) LSTM3: reads [x, (zeroed h1 region), w_new, h2_new, h3_old]
    {
      v8f acc[4] = {{}, {}, {}, {}};
      lstm_mm(W3t, K3, KT3, xin, m, hi, u, acc);
      v8f h;
      lstm_pointwise(acc, c3, bi3, bf3, bg3, bo3, q3i, q3f, q3o, h);
      __syncthreads();
      store_h_lds(xin, SH3, u, hi, h);
#pragma unroll
      for (int r = 0; r < 8; ++r) {    // stream h3 to workspace for MDN GEMM
        int mr = r + hi * 8;
        hseq[((size_t)(row0 + mr) * TSTEPS + t) * UNITS + u] = (_Float16)h[r];
      }
    }
    __syncthreads();
  }
}

// ---- MDN head: (B*T, 400) @ (400,121) + activations ------------------------
__global__ __launch_bounds__(256) void mdn_head(
    const _Float16* __restrict__ hseq, const _Float16* __restrict__ Wt,
    const float* __restrict__ bias, float* __restrict__ out) {
  __shared__ __align__(16) _Float16 aT[16 * KMDN];
  __shared__ float zS[16 * NMDN];

  const int tid = threadIdx.x;
  const int lane = tid & 31;
  const int wid = tid >> 5;           // 0..7: one 16-col tile each
  const int hi = lane >> 4;
  const int m = lane & 15;
  const size_t grow0 = (size_t)blockIdx.x * 16;

  for (int i = tid; i < 16 * KMDN; i += 256) {
    int r = i / KMDN, k = i % KMDN;
    aT[i] = (k < 400) ? hseq[(grow0 + r) * UNITS + k] : (_Float16)0.f;
  }
  __syncthreads();

  const _Float16* wr = Wt + (size_t)(wid * 16 + (lane & 15)) * KMDN + hi * 16;
  v16h bb = *(const v16h*)wr;
  v8f acc = {};
#pragma unroll 1
  for (int kt = 0; kt < 12; ++kt) {
    v16h a = load_a_frag(&aT[m * KMDN + kt * 32], hi * 8);
    v16h bn = *(const v16h*)(wr + (kt + 1) * 32);
    acc = WMMA(a, bb, acc);
    bb = bn;
  }
  {
    v16h a = load_a_frag(&aT[m * KMDN + 12 * 32], hi * 8);
    acc = WMMA(a, bb, acc);
  }
  {
    int col = wid * 16 + (lane & 15);
    float bv = (col < 121) ? bias[col] : 0.f;
#pragma unroll
    for (int r = 0; r < 8; ++r) zS[(r + hi * 8) * NMDN + col] = acc[r] + bv;
  }
  __syncthreads();

  if (tid < 16) {
    float* z = &zS[tid * NMDN];
    float* o = out + (grow0 + tid) * 121;
    float mx = z[0];
    for (int j = 1; j < NMIX; ++j) mx = fmaxf(mx, z[j]);
    float s = 0.f;
    for (int j = 0; j < NMIX; ++j) { z[j] = __expf(z[j] - mx); s += z[j]; }
    float inv = 1.f / s;
    for (int j = 0; j < NMIX; ++j)            o[j] = z[j] * inv;        // pi
    for (int j = NMIX; j < 3 * NMIX; ++j)     o[j] = z[j];              // mu1, mu2
    for (int j = 3 * NMIX; j < 5 * NMIX; ++j) o[j] = __expf(z[j]);      // sig1, sig2
    for (int j = 5 * NMIX; j < 6 * NMIX; ++j) o[j] = tanhf(z[j]);       // rho
    o[6 * NMIX] = sigf(z[6 * NMIX]);                                    // eos
  }
}

extern "C" void kernel_launch(void* const* d_in, const int* in_sizes, int n_in,
                              void* d_out, int out_size, void* d_ws, size_t ws_size,
                              hipStream_t stream) {
  const float* inputs   = (const float*)d_in[0];
  const int*   char_seq = (const int*)d_in[1];
  const int*   char_len = (const int*)d_in[2];
  const float* w1 = (const float*)d_in[3];
  const float* b1 = (const float*)d_in[4];
  const float* p1i = (const float*)d_in[5];
  const float* p1f = (const float*)d_in[6];
  const float* p1o = (const float*)d_in[7];
  const float* w2 = (const float*)d_in[8];
  const float* b2 = (const float*)d_in[9];
  const float* p2i = (const float*)d_in[10];
  const float* p2f = (const float*)d_in[11];
  const float* p2o = (const float*)d_in[12];
  const float* w3 = (const float*)d_in[13];
  const float* b3 = (const float*)d_in[14];
  const float* p3i = (const float*)d_in[15];
  const float* p3f = (const float*)d_in[16];
  const float* p3o = (const float*)d_in[17];
  const float* attk = (const float*)d_in[18];
  const float* attb = (const float*)d_in[19];
  const float* mdnk = (const float*)d_in[20];
  const float* mdnb = (const float*)d_in[21];
  float* out = (float*)d_out;

  // workspace layout (all offsets 256B-aligned)
  char* ws = (char*)d_ws;
  const size_t off_w1 = 0;
  const size_t off_w2 = off_w1 + (size_t)1600 * K1 * 2;   // 1,536,000
  const size_t off_w3 = off_w2 + (size_t)1600 * K2 * 2;   // +2,867,200
  const size_t off_wa = off_w3 + (size_t)1600 * K3 * 2;   // +4,096,000
  const size_t off_wm = off_wa + (size_t)32 * K1 * 2;     // +30,720
  const size_t off_hs = off_wm + (size_t)NMDN * KMDN * 2; // +106,496
  _Float16* W1t  = (_Float16*)(ws + off_w1);
  _Float16* W2t  = (_Float16*)(ws + off_w2);
  _Float16* W3t  = (_Float16*)(ws + off_w3);
  _Float16* Wat  = (_Float16*)(ws + off_wa);
  _Float16* Wmt  = (_Float16*)(ws + off_wm);
  _Float16* hseq = (_Float16*)(ws + off_hs);              // 256*600*400 f16

  conv_w1<<<(1600 * K1 + 255) / 256, 256, 0, stream>>>(w1, W1t);
  conv_w2<<<(1600 * K2 + 255) / 256, 256, 0, stream>>>(w2, W2t);
  conv_w3<<<(1600 * K3 + 255) / 256, 256, 0, stream>>>(w3, W3t);
  conv_wa<<<(32 * K1 + 255) / 256, 256, 0, stream>>>(attk, Wat);
  conv_wm<<<(NMDN * KMDN + 255) / 256, 256, 0, stream>>>(mdnk, Wmt);

  handwriting_recurrent<<<16, 800, 0, stream>>>(
      inputs, char_seq, char_len, W1t, W2t, W3t, Wat,
      b1, p1i, p1f, p1o, b2, p2i, p2f, p2o, b3, p3i, p3f, p3o,
      attb, hseq);

  mdn_head<<<(256 * TSTEPS) / 16, 256, 0, stream>>>(hseq, Wmt, mdnb, out);
}
